// DeformableTransformer_13606456394032
// MI455X (gfx1250) — matrix operand order repaired and checked by reading, not compiled, not run on hardware
//
#include <hip/hip_runtime.h>
#include <hip/hip_bf16.h>

typedef __attribute__((ext_vector_type(16))) _Float16 v16h;
typedef __attribute__((ext_vector_type(8)))  _Float16 v8h;
typedef __attribute__((ext_vector_type(2)))  _Float16 v2h;
typedef __attribute__((ext_vector_type(8)))  float    v8f;
typedef __attribute__((ext_vector_type(4)))  float    v4f;

#define NQC 900
#define TTOT 3840
#define DM 256

union U16 { v16h v; v8h h[2]; };

// ---------------------------------------------------------------------------
// f32->f16 WMMA GEMM: C[M,N] = A[M,K] @ B[K,N] + bias, optional relu.
// REQUIRES: M%16==0, N%64==0 (grid-covered), K%32==0, bias != nullptr.
// Block = 128 threads (4 waves) -> 32(M) x 64(N) tile. K staged in 32-chunks
// through f16 LDS: A as one b128 store/thread, B transposed with paired-K b32
// stores. Fragments read back as ds_load_b128; 2 WMMAs per wave per K-step.
// ---------------------------------------------------------------------------
__global__ __launch_bounds__(128)
void gemm_f16wmma(const float* __restrict__ A, int lda,
                  const float* __restrict__ B, int ldb,
                  const float* __restrict__ bias,
                  float* __restrict__ C, int ldc,
                  int M, int N, int K, int relu) {
  __shared__ _Float16 As[32][40];   // [m][k], 80B rows (16B-aligned)
  __shared__ _Float16 Bs[64][40];   // [n][k]
  int tid = threadIdx.x;
  int wave = tid >> 5, lane = tid & 31, hf = lane >> 4, l16 = lane & 15;
  int tm  = blockIdx.y * 32;
  int tn0 = blockIdx.x * 64;
  bool twoM = (tm + 16 < M);

  int arow = tid >> 2;              // 0..31
  int acol = (tid & 3) * 8;         // 0,8,16,24
  int arow_g = min(tm + arow, M - 1);
  int bkr = (tid >> 3) * 2;         // 0,2,...,30 (K-row pair)
  int bnc = (tid & 7) * 8;          // 0..56 step 8

  v8f acc0 = {}, acc1 = {};
  for (int k0 = 0; k0 < K; k0 += 32) {
    // A tile 32x32: 8 floats/thread -> one ds_store_b128
    {
      const v4f* p = (const v4f*)(A + (size_t)arow_g * lda + k0 + acol);
      v4f x0 = p[0], x1 = p[1];
      v8h hv;
#pragma unroll
      for (int j = 0; j < 4; ++j) {
        hv[j]     = (_Float16)x0[j];
        hv[4 + j] = (_Float16)x1[j];
      }
      *(v8h*)&As[arow][acol] = hv;
    }
    // B tile 32x64 transposed: rows {bkr,bkr+1} x 8 cols; paired-K b32 stores
    {
      const v4f* p0 = (const v4f*)(B + (size_t)(k0 + bkr) * ldb + tn0 + bnc);
      const v4f* p1 = (const v4f*)(B + (size_t)(k0 + bkr + 1) * ldb + tn0 + bnc);
      v4f r0a = p0[0], r0b = p0[1], r1a = p1[0], r1b = p1[1];
#pragma unroll
      for (int j = 0; j < 4; ++j) {
        v2h t0; t0[0] = (_Float16)r0a[j]; t0[1] = (_Float16)r1a[j];
        *(v2h*)&Bs[bnc + j][bkr] = t0;
        v2h t1; t1[0] = (_Float16)r0b[j]; t1[1] = (_Float16)r1b[j];
        *(v2h*)&Bs[bnc + 4 + j][bkr] = t1;
      }
    }
    // prefetch next K-chunk while this one computes (global_prefetch_b8)
    if (k0 + 32 < K) {
      __builtin_prefetch(A + (size_t)arow_g * lda + k0 + 32 + acol, 0, 3);
      __builtin_prefetch(B + (size_t)(k0 + 32 + bkr) * ldb + tn0 + bnc, 0, 3);
    }
    __syncthreads();
    U16 ua0, ua1, ub;
    ua0.h[0] = *(const v8h*)&As[l16][8 * hf];
    ua0.h[1] = *(const v8h*)&As[l16][16 + 8 * hf];
    ua1.h[0] = *(const v8h*)&As[16 + l16][8 * hf];
    ua1.h[1] = *(const v8h*)&As[16 + l16][16 + 8 * hf];
    const _Float16* bp = &Bs[wave * 16 + l16][8 * hf];
    ub.h[0] = *(const v8h*)bp;
    ub.h[1] = *(const v8h*)(bp + 16);
    acc0 = __builtin_amdgcn_wmma_f32_16x16x32_f16(false, ua0.v, false, ub.v,
                                                  (short)0, acc0, false, false);
    acc1 = __builtin_amdgcn_wmma_f32_16x16x32_f16(false, ua1.v, false, ub.v,
                                                  (short)0, acc1, false, false);
    __syncthreads();
  }

  int bcol = tn0 + wave * 16 + l16;
  float bv = bias[bcol];
#pragma unroll
  for (int i = 0; i < 8; ++i) {
    float v = acc0[i] + bv;
    if (relu) v = fmaxf(v, 0.0f);
    C[(size_t)(tm + i + 8 * hf) * ldc + bcol] = v;
  }
  if (twoM) {
#pragma unroll
    for (int i = 0; i < 8; ++i) {
      float v = acc1[i] + bv;
      if (relu) v = fmaxf(v, 0.0f);
      C[(size_t)(tm + 16 + i + 8 * hf) * ldc + bcol] = v;
    }
  }
}

// ---------------------------------------------------------------------------
// Fused flash-style MHA. qkv: [B*NQ, 768] = [Q|K|V], Dh=32, H=8.
// One wave per block = (16-query tile, head, batch). Branch-free clamped
// loads; V tile staged transposed through LDS (b128 in, b128 fragments out).
// ---------------------------------------------------------------------------
__global__ __launch_bounds__(32)
void mha_attn(const float* __restrict__ qkv, float* __restrict__ o) {
  int qt = blockIdx.x, h = blockIdx.y, b = blockIdx.z;
  int lane = threadIdx.x & 31;
  int hf = lane >> 4, l16 = lane & 15;
  __shared__ _Float16 Pl[16][40];    // P tile  [q][k]
  __shared__ _Float16 Vt[32][40];    // V tile transposed [d][k]
  const float scale = 0.17677669529663687f; // 1/sqrt(32)

  // Q fragment (A layout): row=l16, two contiguous 8-float runs
  v16h aq;
  {
    int qn = qt * 16 + l16;
    float qm = (qn < NQC) ? scale : 0.0f;
    const float* qrow =
        qkv + (size_t)(b * NQC + min(qn, NQC - 1)) * 768 + h * 32;
    v4f q0 = ((const v4f*)(qrow + 16 * hf))[0];
    v4f q1 = ((const v4f*)(qrow + 16 * hf))[1];
    v4f q2 = ((const v4f*)(qrow + 16 * hf + 8))[0];
    v4f q3 = ((const v4f*)(qrow + 16 * hf + 8))[1];
#pragma unroll
    for (int j = 0; j < 4; ++j) {
      aq[j]      = (_Float16)(q0[j] * qm);
      aq[4 + j]  = (_Float16)(q1[j] * qm);
      aq[8 + j]  = (_Float16)(q2[j] * qm);
      aq[12 + j] = (_Float16)(q3[j] * qm);
    }
  }

  float m[8], lsum[8];
  v8f o0 = {}, o1 = {};
#pragma unroll
  for (int i = 0; i < 8; ++i) { m[i] = -1e30f; lsum[i] = 0.0f; }

  const int KT = (NQC + 31) / 32; // 29
  for (int kt = 0; kt < KT; ++kt) {
    int k0 = kt * 32;
    // V rows: lane owns key k0+lane; 8x b128 coalesced loads
    int vkey = k0 + lane;
    float vm = (vkey < NQC) ? 1.0f : 0.0f;
    const float* vrow =
        qkv + (size_t)(b * NQC + min(vkey, NQC - 1)) * 768 + 512 + h * 32;
    v4f vv[8];
#pragma unroll
    for (int q = 0; q < 8; ++q) vv[q] = ((const v4f*)vrow)[q];

    // S = Q x K^T (two 16-key subtiles)
    v8f s01[2];
#pragma unroll
    for (int j = 0; j < 2; ++j) {
      int key = k0 + j * 16 + l16;
      float km = (key < NQC) ? 1.0f : 0.0f;
      const float* krow =
          qkv + (size_t)(b * NQC + min(key, NQC - 1)) * 768 + 256 + h * 32 +
          16 * hf;
      v16h bk;
#pragma unroll
      for (int q = 0; q < 4; ++q) {
        v4f kv = ((const v4f*)krow)[q];
#pragma unroll
        for (int j2 = 0; j2 < 4; ++j2) bk[q * 4 + j2] = (_Float16)(kv[j2] * km);
      }
      v8f z = {};
      s01[j] = __builtin_amdgcn_wmma_f32_16x16x32_f16(false, aq, false, bk,
                                                      (short)0, z, false, false);
    }
    // mask invalid key columns (branch-free)
#pragma unroll
    for (int j = 0; j < 2; ++j) {
      bool okc = (k0 + j * 16 + l16) < NQC;
#pragma unroll
      for (int i = 0; i < 8; ++i) s01[j][i] = okc ? s01[j][i] : -1e30f;
    }
    // online softmax (row = i + 8*hf across 16 lanes of each half)
    float alpha[8];
#pragma unroll
    for (int i = 0; i < 8; ++i) {
      float mx = fmaxf(s01[0][i], s01[1][i]);
#pragma unroll
      for (int off = 1; off < 16; off <<= 1)
        mx = fmaxf(mx, __shfl_xor(mx, off, 32));
      float mn = fmaxf(m[i], mx);
      float p0 = __expf(s01[0][i] - mn);
      float p1 = __expf(s01[1][i] - mn);
      float rs = p0 + p1;
#pragma unroll
      for (int off = 1; off < 16; off <<= 1) rs += __shfl_xor(rs, off, 32);
      alpha[i] = __expf(m[i] - mn);
      lsum[i] = lsum[i] * alpha[i] + rs;
      m[i] = mn;
      s01[0][i] = p0; s01[1][i] = p1;
    }
#pragma unroll
    for (int i = 0; i < 8; ++i) { o0[i] *= alpha[i]; o1[i] *= alpha[i]; }

    // stage P (C->A transpose) and V (row->col transpose) through LDS
    __syncthreads();   // previous iteration's fragment reads complete
#pragma unroll
    for (int i = 0; i < 8; ++i) {
      Pl[i + 8 * hf][l16]      = (_Float16)s01[0][i];
      Pl[i + 8 * hf][16 + l16] = (_Float16)s01[1][i];
    }
#pragma unroll
    for (int d = 0; d < 32; ++d)
      Vt[d][lane] = (_Float16)(vv[d >> 2][d & 3] * vm);
    __syncthreads();

    U16 up;
    up.h[0] = *(const v8h*)&Pl[l16][16 * hf];
    up.h[1] = *(const v8h*)&Pl[l16][16 * hf + 8];
#pragma unroll
    for (int j = 0; j < 2; ++j) {
      U16 ubv;
      const _Float16* vp = &Vt[j * 16 + l16][16 * hf];
      ubv.h[0] = *(const v8h*)vp;
      ubv.h[1] = *(const v8h*)(vp + 8);
      if (j == 0)
        o0 = __builtin_amdgcn_wmma_f32_16x16x32_f16(false, up.v, false, ubv.v,
                                                    (short)0, o0, false, false);
      else
        o1 = __builtin_amdgcn_wmma_f32_16x16x32_f16(false, up.v, false, ubv.v,
                                                    (short)0, o1, false, false);
    }
  }

#pragma unroll
  for (int i = 0; i < 8; ++i) {
    int qn = qt * 16 + i + 8 * hf;
    if (qn < NQC) {
      float inv = 1.0f / lsum[i];
      float* orow = o + (size_t)(b * NQC + qn) * DM + h * 32;
      orow[l16] = o0[i] * inv;
      orow[16 + l16] = o1[i] * inv;
    }
  }
}

// ---------------------------------------------------------------------------
// Elementwise / small kernels
// ---------------------------------------------------------------------------
__global__ void add_k(const float* a, const float* b, float* dst, int n) {
  int i = blockIdx.x * blockDim.x + threadIdx.x;
  if (i < n) dst[i] = a[i] + b[i];
}

__global__ void mask_k(float* v, const unsigned char* mask, int n) {
  int i = blockIdx.x * blockDim.x + threadIdx.x;
  if (i < n) { float z = mask[i >> 8] ? 0.0f : v[i]; v[i] = z; }
}

__global__ void layernorm_res(const float* __restrict__ x,
                              const float* __restrict__ y,
                              const float* __restrict__ g,
                              const float* __restrict__ bta,
                              float* __restrict__ dst) {
  int t = blockIdx.x;
  int lane = threadIdx.x;
  float vals[8];
  float s = 0.f, sq = 0.f;
#pragma unroll
  for (int j = 0; j < 8; ++j) {
    int c = lane + j * 32;
    float v = x[(size_t)t * DM + c] + y[(size_t)t * DM + c];
    vals[j] = v; s += v; sq += v * v;
  }
#pragma unroll
  for (int off = 1; off < 32; off <<= 1) {
    s += __shfl_xor(s, off, 32);
    sq += __shfl_xor(sq, off, 32);
  }
  float mean = s * (1.0f / DM);
  float var = sq * (1.0f / DM) - mean * mean;
  float r = rsqrtf(var + 1e-5f);
#pragma unroll
  for (int j = 0; j < 8; ++j) {
    int c = lane + j * 32;
    dst[(size_t)t * DM + c] = (vals[j] - mean) * r * g[c] + bta[c];
  }
}

__global__ void ref_kernel(const float* seg, const float* dur, const float* vr,
                           float* ref, float* refin) {
  int t = blockIdx.x * blockDim.x + threadIdx.x;
  if (t >= 4 * NQC) return;
  int b = t / NQC;
  float c = seg[(size_t)t * 2];
  float w = __expf(seg[(size_t)t * 2 + 1]);
  ref[(size_t)t * 2] = c;
  ref[(size_t)t * 2 + 1] = w;
  float inv = 1.0f / dur[b];
  for (int l = 0; l < 4; ++l) {
    float r = vr[b * 4 + l];
    refin[((size_t)t * 4 + l) * 2]     = c * inv * r;
    refin[((size_t)t * 4 + l) * 2 + 1] = w * inv * r;
  }
}

__global__ void pos_kernel(const float* ref, float* sine) {
  int i = blockIdx.x * blockDim.x + threadIdx.x; // t*512 + j
  if (i >= 4 * NQC * 512) return;
  int t = i >> 9, j = i & 511;
  int part = j >> 8, jj = j & 255;
  float p = ref[(size_t)t * 2 + part] * 6.283185307179586f;
  float e = (float)(2 * (jj >> 1)) * (1.0f / 256.0f);
  float v = p / __powf(10000.0f, e);
  sine[i] = (jj & 1) ? __cosf(v) : __sinf(v);
}

__global__ void softmax16(float* a, int n) {
  int i = blockIdx.x * blockDim.x + threadIdx.x;
  if (i >= n) return;
  float* p = a + (size_t)i * 16;
  float mx = -1e30f;
#pragma unroll
  for (int j = 0; j < 16; ++j) mx = fmaxf(mx, p[j]);
  float e[16]; float s = 0.f;
#pragma unroll
  for (int j = 0; j < 16; ++j) { e[j] = __expf(p[j] - mx); s += e[j]; }
  float inv = 1.0f / s;
#pragma unroll
  for (int j = 0; j < 16; ++j) p[j] = e[j] * inv;
}

__global__ void loc_kernel(const float* off, const float* refin, float* loc,
                           int boundary) {
  int i = blockIdx.x * blockDim.x + threadIdx.x; // t*128 + c
  if (i >= 4 * NQC * 128) return;
  int t = i >> 7, c = i & 127;
  int l = (c >> 2) & 3, p = c & 3;
  float center = refin[((size_t)t * 4 + l) * 2];
  float width  = refin[((size_t)t * 4 + l) * 2 + 1];
  float base = center;
  if (boundary) base = (p < 2) ? center - width * 0.5f : center + width * 0.5f;
  loc[i] = base + off[i] * 0.25f * width * 0.5f;
}

__global__ void dsample(const float* __restrict__ value,
                        const float* __restrict__ loc,
                        const float* __restrict__ aw,
                        const long long* __restrict__ lens,
                        const long long* __restrict__ starts,
                        float* __restrict__ outp) {
  int i = blockIdx.x * blockDim.x + threadIdx.x; // t*256 + c
  if (i >= 4 * NQC * DM) return;
  int t = i >> 8, c = i & 255;
  int h = c >> 5;
  int b = t / NQC;
  float acc = 0.0f;
  for (int l = 0; l < 4; ++l) {
    int Tl = (int)lens[l];
    int s  = (int)starts[l];
    const float* vbase = value + ((size_t)b * TTOT + s) * DM + c;
    for (int p = 0; p < 4; ++p) {
      float lx = loc[(size_t)t * 128 + h * 16 + l * 4 + p];
      float x = lx * (float)Tl - 0.5f;
      float fl = floorf(x);
      float w = x - fl;
      int i0 = (int)fl, i1 = i0 + 1;
      int i0c = min(max(i0, 0), Tl - 1);
      int i1c = min(max(i1, 0), Tl - 1);
      float ok0 = (i0 >= 0 && i0 < Tl) ? 1.0f : 0.0f;
      float ok1 = (i1 >= 0 && i1 < Tl) ? 1.0f : 0.0f;
      float v0 = vbase[(size_t)i0c * DM];
      float v1 = vbase[(size_t)i1c * DM];
      acc += (v0 * (1.0f - w) * ok0 + v1 * w * ok1) *
             aw[(size_t)t * 128 + h * 16 + l * 4 + p];
    }
  }
  outp[i] = acc;
}

// ---------------------------------------------------------------------------
// Host side
// ---------------------------------------------------------------------------
struct Lin { const float* w; const float* b; };
struct LN  { const float* g; const float* b; };
struct Deform { Lin so, aw, v, o; };
struct Layer {
  Lin sa_in, sa_out; Deform ca, ca_se;
  LN n1, n2, nse, n3; Lin l1, l2;
};

extern "C" void kernel_launch(void* const* d_in, const int* in_sizes, int n_in,
                              void* d_out, int out_size, void* d_ws, size_t ws_size,
                              hipStream_t stream) {
  (void)in_sizes; (void)n_in; (void)out_size; (void)ws_size;
  const int M = 4 * NQC;            // 3600 tokens
  const float* tgt   = (const float*)d_in[0];
  const float* segin = (const float*)d_in[1];
  const float* dur   = (const float*)d_in[2];
  const float* src   = (const float*)d_in[4];
  const long long* lens   = (const long long*)d_in[5];
  const long long* starts = (const long long*)d_in[6];
  const float* vr    = (const float*)d_in[7];
  const unsigned char* pmask = (const unsigned char*)d_in[8];

  // params flattened in jax tree order (sorted dict keys)
  int pi = 9;
  auto nxt = [&]() { return (const float*)d_in[pi++]; };
  Lin gh1, gh2;
  gh1.b = nxt(); gh1.w = nxt();
  gh2.b = nxt(); gh2.w = nxt();
  Layer L[6];
  for (int i = 0; i < 6; ++i) {
    auto rdDef = [&](Deform& d) {
      d.aw.b = nxt(); d.aw.w = nxt();
      d.o.b  = nxt(); d.o.w  = nxt();
      d.so.b = nxt(); d.so.w = nxt();
      d.v.b  = nxt(); d.v.w  = nxt();
    };
    rdDef(L[i].ca); rdDef(L[i].ca_se);
    L[i].l1.b = nxt(); L[i].l1.w = nxt();
    L[i].l2.b = nxt(); L[i].l2.w = nxt();
    L[i].n1.b = nxt(); L[i].n1.g = nxt();
    L[i].n2.b = nxt(); L[i].n2.g = nxt();
    L[i].n3.b = nxt(); L[i].n3.g = nxt();
    L[i].nse.b = nxt(); L[i].nse.g = nxt();
    L[i].sa_in.b = nxt(); L[i].sa_in.w = nxt();
    L[i].sa_out.b = nxt(); L[i].sa_out.w = nxt();
  }

  // workspace bump allocator (floats)
  float* ws = (float*)d_ws;
  size_t cur = 0;
  auto alloc = [&](size_t n) { float* p = ws + cur; cur += n; return p; };
  float* refb   = alloc((size_t)M * 2);
  float* refin  = alloc((size_t)M * 8);
  float* sine   = alloc((size_t)M * 512);
  float* h1     = alloc((size_t)M * DM);
  float* qpos   = alloc((size_t)M * DM);
  float* outbuf = alloc((size_t)M * DM);
  float* q2     = alloc((size_t)M * DM);
  float* qkv    = alloc((size_t)M * 768);
  float* attnout= alloc((size_t)M * DM);
  float* t2     = alloc((size_t)M * DM);
  float* value  = alloc((size_t)4 * TTOT * DM);
  float* offb   = alloc((size_t)M * 128);
  float* awb    = alloc((size_t)M * 128);
  float* locb   = alloc((size_t)M * 128);
  float* dattn  = alloc((size_t)M * DM);
  float* ffh    = alloc((size_t)M * 1024);

  float* outF = (float*)d_out;
  float* out_tok   = outF;                // [4,900,256]
  float* out_seg   = outF + 921600;       // [4,900,2]
  float* out_loc   = outF + 928800;       // [4,900,8,4,4]
  float* out_locse = outF + 1389600;
  float* out_off   = outF + 1850400;
  float* out_offse = outF + 2311200;

  auto gemm = [&](const float* A, int lda, const float* Bm, int ldb,
                  const float* bias, float* C, int ldc,
                  int Mm, int Nn, int Kk, int relu) {
    dim3 g((Nn + 63) / 64, (Mm + 31) / 32);
    gemm_f16wmma<<<g, dim3(128), 0, stream>>>(A, lda, Bm, ldb, bias, C, ldc,
                                              Mm, Nn, Kk, relu);
  };

  // ---- once-per-forward prologue ----
  hipMemcpyAsync(out_seg, segin, (size_t)M * 2 * sizeof(float),
                 hipMemcpyDeviceToDevice, stream);
  ref_kernel<<<(M + 255) / 256, 256, 0, stream>>>(segin, dur, vr, refb, refin);
  pos_kernel<<<((size_t)M * 512 + 255) / 256, 256, 0, stream>>>(refb, sine);
  gemm(sine, 512, gh1.w, DM, gh1.b, h1, DM, M, DM, 512, 1);
  gemm(h1, DM, gh2.w, DM, gh2.b, qpos, DM, M, DM, DM, 0);
  hipMemcpyAsync(outbuf, tgt, (size_t)M * DM * sizeof(float),
                 hipMemcpyDeviceToDevice, stream);

  const int NELE = M * DM;
  auto deform = [&](const Deform& dp, int boundary, float* locdst,
                    float* offdst, const LN& nrm) {
    add_k<<<(NELE + 255) / 256, 256, 0, stream>>>(outbuf, qpos, q2, NELE);
    gemm(src, DM, dp.v.w, DM, dp.v.b, value, DM, 4 * TTOT, DM, DM, 0);
    mask_k<<<((size_t)4 * TTOT * DM + 255) / 256, 256, 0, stream>>>(
        value, pmask, 4 * TTOT * DM);
    gemm(q2, DM, dp.so.w, 128, dp.so.b, offb, 128, M, 128, DM, 0);
    gemm(q2, DM, dp.aw.w, 128, dp.aw.b, awb, 128, M, 128, DM, 0);
    softmax16<<<(M * 8 + 255) / 256, 256, 0, stream>>>(awb, M * 8);
    loc_kernel<<<(M * 128 + 255) / 256, 256, 0, stream>>>(offb, refin, locdst,
                                                          boundary);
    if (offdst)
      hipMemcpyAsync(offdst, offb, (size_t)M * 128 * sizeof(float),
                     hipMemcpyDeviceToDevice, stream);
    dsample<<<(NELE + 255) / 256, 256, 0, stream>>>(value, locdst, awb, lens,
                                                    starts, dattn);
    gemm(dattn, DM, dp.o.w, DM, dp.o.b, t2, DM, M, DM, DM, 0);
    layernorm_res<<<M, 32, 0, stream>>>(outbuf, t2, nrm.g, nrm.b, outbuf);
  };

  for (int li = 0; li < 6; ++li) {
    Layer& lp = L[li];
    bool last = (li == 5);
    // self-attention: q = out + qpos (Q,K); V from out
    add_k<<<(NELE + 255) / 256, 256, 0, stream>>>(outbuf, qpos, q2, NELE);
    gemm(q2, DM, lp.sa_in.w, 768, lp.sa_in.b, qkv, 768, M, 512, DM, 0);
    gemm(outbuf, DM, lp.sa_in.w + 512, 768, lp.sa_in.b + 512, qkv + 512, 768,
         M, DM, DM, 0);
    mha_attn<<<dim3((NQC + 15) / 16, 8, 4), 32, 0, stream>>>(qkv, attnout);
    gemm(attnout, DM, lp.sa_out.w, DM, lp.sa_out.b, t2, DM, M, DM, DM, 0);
    layernorm_res<<<M, 32, 0, stream>>>(outbuf, t2, lp.n2.g, lp.n2.b, outbuf);
    // deformable cross-attention (center-based), then boundary-aware variant
    deform(lp.ca, 0, last ? out_loc : locb, last ? out_off : nullptr, lp.n1);
    deform(lp.ca_se, 1, last ? out_locse : locb, last ? out_offse : nullptr,
           lp.nse);
    // FFN
    gemm(outbuf, DM, lp.l1.w, 1024, lp.l1.b, ffh, 1024, M, 1024, DM, 1);
    gemm(ffh, 1024, lp.l2.w, DM, lp.l2.b, t2, DM, M, DM, 1024, 0);
    layernorm_res<<<M, 32, 0, stream>>>(outbuf, t2, lp.n3.g, lp.n3.b, outbuf);
  }

  hipMemcpyAsync(out_tok, outbuf, (size_t)M * DM * sizeof(float),
                 hipMemcpyDeviceToDevice, stream);
}
